// Attention_24902220382268
// MI455X (gfx1250) — matrix operand order repaired and checked by reading, not compiled, not run on hardware
//
#include <hip/hip_runtime.h>

typedef __attribute__((ext_vector_type(16))) _Float16 v16h;
typedef __attribute__((ext_vector_type(8)))  _Float16 v8h;
typedef __attribute__((ext_vector_type(8)))  float    v8f;

#define WMMA(a, b, c) __builtin_amdgcn_wmma_f32_16x16x32_f16( \
    false, (a), false, (b), (short)0, (c), false, false)

constexpr int   Bn = 8, Cc = 128, Nn = 4096;
constexpr float INV_TEMP = 0.0883883476483184f; // 1/sqrt(128)

// ---------------------------------------------------------------------------
// Kernel 0: convert the four 128x128 fp32 weight matrices to fp16 (row major)
// ---------------------------------------------------------------------------
__global__ __launch_bounds__(256) void cvt_w_kernel(
    const float* __restrict__ wq, const float* __restrict__ wk,
    const float* __restrict__ wv, const float* __restrict__ wo,
    _Float16* __restrict__ wh) {
  int i = blockIdx.x * 256 + threadIdx.x;   // 0..16383
  wh[0 * 16384 + i] = (_Float16)wq[i];
  wh[1 * 16384 + i] = (_Float16)wk[i];
  wh[2 * 16384 + i] = (_Float16)wv[i];
  wh[3 * 16384 + i] = (_Float16)wo[i];
}

// ---------------------------------------------------------------------------
// Kernel 1: fused QKV projection.
//   q[b][n][d] = (Wq x + bq) / temp   (fp16)
//   k[b][n][d] =  Wk x + bk           (fp16)
//   v[b][d][n] =  Wv x + bv           (fp16, transposed for PV A-tiles)
// One block = one batch x 128-column (n) slab; 8 waves, 16 rows (n) each.
// ---------------------------------------------------------------------------
__global__ __launch_bounds__(256) void qkv_proj_kernel(
    const float* __restrict__ x,
    const float* __restrict__ bq, const float* __restrict__ bk,
    const float* __restrict__ bv,
    const _Float16* __restrict__ wh,     // [3][128][128] fp16 (q,k,v)
    _Float16* __restrict__ q, _Float16* __restrict__ k,
    _Float16* __restrict__ v) {
  constexpr int XS = 144;                           // halves per row (16B-aligned rows)
  __shared__ __align__(16) _Float16 xs[128 * XS];   // xT tile: [n_local][c] fp16

  const int blk  = blockIdx.x;
  const int b    = blk >> 5;
  const int n0   = (blk & 31) * 128;
  const int tid  = threadIdx.x;
  const int lane = tid & 31;
  const int w    = tid >> 5;
  const int m    = lane & 15;
  const int h    = lane >> 4;

  // Stage + transpose + convert: x[b][c][n0+nl] (f32) -> xs[nl][c] (f16)
  {
    const int nl = tid & 127;
    const int cb = tid >> 7;   // 0..1
    const float* xp = x + (size_t)b * Cc * Nn + n0 + nl;
#pragma unroll 4
    for (int cc = 0; cc < 64; ++cc) {
      int c = cc * 2 + cb;
      xs[nl * XS + c] = (_Float16)xp[(size_t)c * Nn];
    }
  }
  __syncthreads();

  // A-tiles for this wave's 16 rows (n), K = c in 4 steps of 32
  const int row = w * 16 + m;
  v16h a[4];
#pragma unroll
  for (int kk = 0; kk < 4; ++kk) {
    const _Float16* base = &xs[row * XS + kk * 32 + h * 16];
    v8h lo = *(const v8h*)(base);
    v8h hi = *(const v8h*)(base + 8);
#pragma unroll
    for (int t = 0; t < 8; ++t) { a[kk][t] = lo[t]; a[kk][8 + t] = hi[t]; }
  }

  const float* biases[3] = {bq, bk, bv};
#pragma unroll 1
  for (int p = 0; p < 3; ++p) {
    const _Float16* wp = wh + p * 16384;
    const float*    bp = biases[p];
#pragma unroll 1
    for (int db = 0; db < 8; ++db) {
      const int d0 = db * 16;
      v8f acc = {};
#pragma unroll
      for (int kk = 0; kk < 4; ++kk) {
        // B[k=c][n=d] = w[d][c]: 32B contiguous per lane
        v16h bt = *(const v16h*)(wp + (size_t)(d0 + m) * Cc + kk * 32 + h * 16);
        acc = WMMA(a[kk], bt, acc);
      }
      const float bias = bp[d0 + m];   // column d of this lane
      if (p == 0) {
#pragma unroll
        for (int r = 0; r < 8; ++r) {
          int nr = n0 + w * 16 + r + 8 * h;
          q[((size_t)b * Nn + nr) * Cc + d0 + m] =
              (_Float16)((acc[r] + bias) * INV_TEMP);
        }
      } else if (p == 1) {
#pragma unroll
        for (int r = 0; r < 8; ++r) {
          int nr = n0 + w * 16 + r + 8 * h;
          k[((size_t)b * Nn + nr) * Cc + d0 + m] = (_Float16)(acc[r] + bias);
        }
      } else {
#pragma unroll
        for (int r = 0; r < 8; ++r) {
          int nr = n0 + w * 16 + r + 8 * h;
          v[((size_t)b * Cc + d0 + m) * Nn + nr] = (_Float16)(acc[r] + bias);
        }
      }
    }
  }
}

// ---------------------------------------------------------------------------
// Kernel 2: transposed flash attention + output projection + residual.
// One block = (batch, 64 rows of i); 4 *independent* waves, 16 cols (i) each.
// S^T = K Q^T  (K = A-tiles, Q = loop-invariant B-tiles)
// O^T += V^T P^T  (V^T = A-tiles; P^T built from S^T registers with a single
//                  cross-half shfl_xor(16) per element -- no LDS, no barriers)
// Softmax runs over rows (j) => per-lane-scalar m/l state, 2 shuffles/step.
// ---------------------------------------------------------------------------
__global__ __launch_bounds__(128) void attn_kernel(
    const float* __restrict__ x,
    const _Float16* __restrict__ qh, const _Float16* __restrict__ kh,
    const _Float16* __restrict__ vh, const _Float16* __restrict__ woh,
    const float* __restrict__ bo, float* __restrict__ out) {
  const int b    = blockIdx.x >> 6;
  const int i0   = (blockIdx.x & 63) * 64;
  const int lane = threadIdx.x & 31;
  const int w    = threadIdx.x >> 5;
  const int m    = lane & 15;
  const int h    = lane >> 4;
  const int iw   = i0 + w * 16;   // this wave's 16 columns (i)

  // Q as loop-invariant B-tiles: lane owns column i = iw+m; 32B contiguous
  v16h qb[4];
  {
    const _Float16* qp = qh + ((size_t)b * Nn + iw + m) * Cc + h * 16;
#pragma unroll
    for (int kk = 0; kk < 4; ++kk) qb[kk] = *(const v16h*)(qp + kk * 32);
  }

  v8f o[8];                       // O^T accumulators: 8 d-blocks x (16d x 16i)
#pragma unroll
  for (int db = 0; db < 8; ++db) o[db] = {};
  float mprev = -1e30f, l = 0.0f; // per-column softmax state (one scalar/lane)

  const _Float16* kbb = kh + (size_t)b * Nn * Cc;
  const _Float16* vbb = vh + (size_t)b * Cc * Nn;

#pragma unroll 1
  for (int j0 = 0; j0 < Nn; j0 += 32) {
    __builtin_prefetch(kbb + (size_t)(j0 + 32 + m) * Cc, 0, 1);
    __builtin_prefetch(vbb + (size_t)m * Nn + j0 + 32, 0, 1);

    // S^T tiles: rows j0..j0+15 (st0) and j0+16..j0+31 (st1), cols i
    v8f st0 = {}, st1 = {};
#pragma unroll
    for (int kk = 0; kk < 4; ++kk) {
      const _Float16* k0 = kbb + (size_t)(j0 + m) * Cc + kk * 32 + h * 16;
      const _Float16* k1 = kbb + (size_t)(j0 + 16 + m) * Cc + kk * 32 + h * 16;
      v8h lo0 = *(const v8h*)k0, hi0 = *(const v8h*)(k0 + 8);
      v8h lo1 = *(const v8h*)k1, hi1 = *(const v8h*)(k1 + 8);
      v16h a0, a1;
#pragma unroll
      for (int t = 0; t < 8; ++t) {
        a0[t] = lo0[t]; a0[8 + t] = hi0[t];
        a1[t] = lo1[t]; a1[8 + t] = hi1[t];
      }
      st0 = WMMA(a0, qb[kk], st0);
      st1 = WMMA(a1, qb[kk], st1);
    }

    // Online softmax over j: local reduce over the 16 register-resident rows,
    // one cross-half shuffle to combine, all state per-lane scalar.
    float mx = fmaxf(st0[0], st1[0]);
#pragma unroll
    for (int r = 1; r < 8; ++r) mx = fmaxf(mx, fmaxf(st0[r], st1[r]));
    mx = fmaxf(mx, __shfl_xor(mx, 16, 32));
    const float mn = fmaxf(mprev, mx);
    const float sc = __expf(mprev - mn);
    float p0[8], p1[8], rs = 0.0f;
#pragma unroll
    for (int r = 0; r < 8; ++r) {
      p0[r] = __expf(st0[r] - mn);
      p1[r] = __expf(st1[r] - mn);
      rs += p0[r] + p1[r];
    }
    rs += __shfl_xor(rs, 16, 32);
    l = l * sc + rs;
    mprev = mn;
#pragma unroll
    for (int db = 0; db < 8; ++db)
#pragma unroll
      for (int r = 0; r < 8; ++r) o[db][r] *= sc;

    // P^T B-operand: C/D layout -> B layout needs only a half-swap per element
    v16h pb;
#pragma unroll
    for (int t = 0; t < 8; ++t) {
      float q0 = __shfl_xor(p0[t], 16, 32);
      float q1 = __shfl_xor(p1[t], 16, 32);
      pb[t]     = (_Float16)(h ? q1 : p0[t]);   // k = h*16 + t
      pb[8 + t] = (_Float16)(h ? p1[t] : q0);   // k = h*16 + 8 + t
    }

    // O^T += V^T * P^T  (V^T rows d as A-tiles, contiguous from v[b][d][n])
#pragma unroll
    for (int db = 0; db < 8; ++db) {
      const _Float16* vp = vbb + (size_t)(db * 16 + m) * Nn + j0 + h * 16;
      v8h lo = *(const v8h*)vp, hi = *(const v8h*)(vp + 8);
      v16h av;
#pragma unroll
      for (int t = 0; t < 8; ++t) { av[t] = lo[t]; av[8 + t] = hi[t]; }
      o[db] = WMMA(av, pb, o[db]);
    }
  }

  // Normalize: y^T = O^T / l (l is lane-uniform per column)
  const float inv = 1.0f / l;
#pragma unroll
  for (int db = 0; db < 8; ++db)
#pragma unroll
    for (int r = 0; r < 8; ++r) o[db][r] *= inv;

  // Build the 4 y^T B-tiles once (register half-swap, same trick as P^T)
  v16h yb[4];
#pragma unroll
  for (int kk = 0; kk < 4; ++kk) {
    const int e = 2 * kk, g = 2 * kk + 1;
#pragma unroll
    for (int t = 0; t < 8; ++t) {
      float s0 = __shfl_xor(o[e][t], 16, 32);
      float s1 = __shfl_xor(o[g][t], 16, 32);
      yb[kk][t]     = (_Float16)(h ? s1 : o[e][t]);
      yb[kk][8 + t] = (_Float16)(h ? o[g][t] : s0);
    }
  }

  // out = x + Wo y + bo; Wo rows (do) as A-tiles from global fp16
#pragma unroll 1
  for (int db = 0; db < 8; ++db) {
    const int d0 = db * 16;
    v8f acc = {};
#pragma unroll
    for (int kk = 0; kk < 4; ++kk) {
      const _Float16* ab = woh + (size_t)(d0 + m) * Cc + kk * 32 + h * 16;
      v8h lo = *(const v8h*)ab, hi = *(const v8h*)(ab + 8);
      v16h at;
#pragma unroll
      for (int t = 0; t < 8; ++t) { at[t] = lo[t]; at[8 + t] = hi[t]; }
      acc = WMMA(at, yb[kk], acc);
    }
#pragma unroll
    for (int r = 0; r < 8; ++r) {
      int    doi = d0 + r + 8 * h;
      size_t idx = ((size_t)b * Cc + doi) * Nn + iw + m;
      out[idx] = x[idx] + acc[r] + bo[doi];
    }
  }
}

// ---------------------------------------------------------------------------
extern "C" void kernel_launch(void* const* d_in, const int* in_sizes, int n_in,
                              void* d_out, int out_size, void* d_ws,
                              size_t ws_size, hipStream_t stream) {
  const float* x  = (const float*)d_in[0];
  const float* wq = (const float*)d_in[1];
  const float* bq = (const float*)d_in[2];
  const float* wk = (const float*)d_in[3];
  const float* bk = (const float*)d_in[4];
  const float* wv = (const float*)d_in[5];
  const float* bv = (const float*)d_in[6];
  const float* wo = (const float*)d_in[7];
  const float* bo = (const float*)d_in[8];

  _Float16* ws = (_Float16*)d_ws;
  _Float16* qh = ws;                               // 8 MB
  _Float16* kh = ws + (size_t)Bn * Nn * Cc;        // 8 MB
  _Float16* vh = ws + 2 * (size_t)Bn * Nn * Cc;    // 8 MB
  _Float16* wh = ws + 3 * (size_t)Bn * Nn * Cc;    // 4 * 32 KB fp16 weights

  cvt_w_kernel<<<64, 256, 0, stream>>>(wq, wk, wv, wo, wh);
  qkv_proj_kernel<<<Bn * (Nn / 128), 256, 0, stream>>>(x, bq, bk, bv, wh, qh,
                                                       kh, vh);
  attn_kernel<<<Bn * (Nn / 64), 128, 0, stream>>>(
      x, qh, kh, vh, wh + 3 * 16384, bo, (float*)d_out);
}
